// EEGTokenizer_46213848105950
// MI455X (gfx1250) — matrix operand order repaired and checked by reading, not compiled, not run on hardware
//
#include <hip/hip_runtime.h>

// ---------- WMMA types / helpers (CDNA5 gfx1250, wave32) ----------
typedef __bf16 v16bf __attribute__((ext_vector_type(16)));
typedef float  v8f   __attribute__((ext_vector_type(8)));

union Frag {
  v16bf v;
  unsigned int u[8];
};

__device__ __forceinline__ unsigned short f2bf(float f) {
  unsigned int u = __float_as_uint(f);
  u += 0x7FFFu + ((u >> 16) & 1u);           // round-to-nearest-even
  return (unsigned short)(u >> 16);
}
__device__ __forceinline__ float bf2f(unsigned short h) {
  return __uint_as_float(((unsigned int)h) << 16);
}
__device__ __forceinline__ v8f zero8() {
  v8f z = {0.f, 0.f, 0.f, 0.f, 0.f, 0.f, 0.f, 0.f};
  return z;
}

// A-matrix 16x32 bf16 fragment (ISA 7.12.2): lane m=l&15, half h=l>>4.
// VGPR j holds K pair: k = (j>>2)*16 + h*8 + (j&3)*2.  base pre-offset to (m0,k0).
__device__ __forceinline__ void load_a(Frag& f, const unsigned short* base, int stride, int lane) {
  const int m = lane & 15, h = (lane >> 4) & 1;
  const unsigned short* p = base + m * stride;
#pragma unroll
  for (int j = 0; j < 8; ++j) {
    const int k = ((j >> 2) << 4) + (h << 3) + ((j & 3) << 1);
    f.u[j] = *(const unsigned int*)(p + k);
  }
}
// B-matrix 32x16 bf16 fragment: lane n=l&15, half h=l>>4 (same K order as A).
// VGPR j holds K pair: k = (j>>2)*16 + h*8 + (j&3)*2.  src layout [n][k] row-major, base pre-offset to (n0,k0).
__device__ __forceinline__ void load_b(Frag& f, const unsigned short* base, int stride, int lane) {
  const int n = lane & 15, h = (lane >> 4) & 1;
  const unsigned short* p = base + n * stride;
#pragma unroll
  for (int j = 0; j < 8; ++j) f.u[j] = *(const unsigned int*)(p + ((j >> 2) << 4) + (h << 3) + ((j & 3) << 1));
}

__device__ __forceinline__ v8f wmma_bf16(const Frag& a, const Frag& b, v8f c) {
  return __builtin_amdgcn_wmma_f32_16x16x32_bf16(false, a.v, false, b.v, (short)0, c, false, false);
}

// Generic K-loop: acc[NT] (16xK row tile x NT 16-col tiles). A pre-offset to (row0,0).
template <int NT>
__device__ __forceinline__ void gemm_acc(v8f* acc, const unsigned short* A, int lda,
                                         const unsigned short* B, int ldb, int ksteps, int lane) {
  for (int ks = 0; ks < ksteps; ++ks) {
    Frag a;
    load_a(a, A + ks * 32, lda, lane);
#pragma unroll
    for (int nt = 0; nt < NT; ++nt) {
      Frag b;
      load_b(b, B + (nt * 16) * ldb + ks * 32, ldb, lane);
      acc[nt] = wmma_bf16(a, b, acc[nt]);
    }
  }
}

// ---------------- prep kernels (weight packing, BN folding) ----------------
__global__ void k_prep_pw(const float* w_sp, const float* b_sp, const float* g1,
                          const float* be1, const float* m1, const float* v1,
                          unsigned short* w1p, float* b1f) {
  int i = blockIdx.x * 256 + threadIdx.x;
  if (i < 64 * 256) {
    int oc = i >> 8;
    float s = g1[oc] * rsqrtf(v1[oc] + 1e-5f);
    w1p[i] = f2bf(w_sp[i] * s);          // [oc][c] bf16, BN1 scale folded
  }
  if (i < 64) {
    float s = g1[i] * rsqrtf(v1[i] + 1e-5f);
    b1f[i] = (b_sp[i] - m1[i]) * s + be1[i];
  }
}

__global__ void k_prep_wt(const float* wt3, const float* wt7, const float* wt15, const float* wt31,
                          unsigned short* wtp) {
  int i = blockIdx.x * 256 + threadIdx.x;
  if (i >= 56 * 4096) return;
  int tg = i >> 12, rem = i & 4095;
  int oc = rem >> 6, ic = rem & 63;
  const float* src; int tl, nt;
  if (tg < 3)       { src = wt3;  tl = tg;      nt = 3;  }
  else if (tg < 10) { src = wt7;  tl = tg - 3;  nt = 7;  }
  else if (tg < 25) { src = wt15; tl = tg - 10; nt = 15; }
  else              { src = wt31; tl = tg - 25; nt = 31; }
  // pack [tap][oc][ic] so B-frag K-pairs (ic,ic+1) are contiguous
  wtp[(tg * 64 + oc) * 64 + ic] = f2bf(src[(oc * 64 + ic) * nt + tl]);
}

__global__ void k_prep_F(unsigned short* F) {
  int i = blockIdx.x * 256 + threadIdx.x;
  if (i >= 112 * 256) return;
  int m = i >> 8, t = i & 255;
  float v = 0.f;
  if (m < 100) {
    int k = (m >> 1) + 1;                 // freq bins 1..50
    int ph = (k * t) & 255;               // exact phase mod 256
    float ang = 6.283185307179586f * (float)ph * (1.0f / 256.0f);
    v = (m & 1) ? __sinf(ang) : __cosf(ang);
  }
  F[m * 256 + t] = f2bf(v);
}

// dst[n*K + k] = bf16(src[k*N + n])  (pack matmul weights as [n][k])
__global__ void k_tpack(const float* src, unsigned short* dst, int K, int N) {
  int i = blockIdx.x * 256 + threadIdx.x;
  if (i >= K * N) return;
  int n = i / K, k = i - n * K;
  dst[n * K + k] = f2bf(src[k * N + n]);
}

__global__ void k_zero_hpad(unsigned short* h) {
  int i = blockIdx.x * 256 + threadIdx.x;
  if (i < 32 * 1280) h[992 * 1280 + i] = 0;  // zero padded M rows of h
}

// ---------------- 1) pointwise conv + BN1 + ReLU : y1 = relu(x @ W1' + b1') ----------------
__global__ __launch_bounds__(256) void k_pointwise(const float* x, const unsigned short* w1p,
                                                   const float* b1f, unsigned short* y1) {
  __shared__ unsigned short Ws[64 * 258];   // weights [oc][c], padded
  __shared__ unsigned short Xs[128 * 34];   // 128 rows x 32 K slice, padded
  const int tid = threadIdx.x, lane = tid & 31, w = tid >> 5;
  const int R0 = blockIdx.x * 128;
  for (int i = tid; i < 64 * 256; i += 256) Ws[(i >> 8) * 258 + (i & 255)] = w1p[i];
  v8f acc[4];
#pragma unroll
  for (int nt = 0; nt < 4; ++nt) acc[nt] = zero8();
  for (int ks = 0; ks < 8; ++ks) {
    __syncthreads();
    for (int i = tid; i < 4096; i += 256) {
      int r = i >> 5, c = i & 31;
      Xs[r * 34 + c] = f2bf(x[(R0 + r) * 256 + ks * 32 + c]);
    }
    __syncthreads();
    Frag a;
    load_a(a, Xs + (w * 16) * 34, 34, lane);
#pragma unroll
    for (int nt = 0; nt < 4; ++nt) {
      Frag b;
      load_b(b, Ws + (nt * 16) * 258 + ks * 32, 258, lane);
      acc[nt] = wmma_bf16(a, b, acc[nt]);
    }
  }
  const int n = lane & 15, hh = lane >> 4;
#pragma unroll
  for (int nt = 0; nt < 4; ++nt) {
    int oc = nt * 16 + n;
    float bb = b1f[oc];
#pragma unroll
    for (int r = 0; r < 8; ++r) {
      int row = R0 + w * 16 + r + 8 * hh;
      float v = acc[nt][r] + bb;
      y1[row * 64 + oc] = f2bf(v > 0.f ? v : 0.f);
    }
  }
}

// ---------------- 2) temporal convs (per-branch) + ReLU + BN2 -> xc bf16 ----------------
// Input tile staged with CDNA5 async global->LDS DMA (ASYNCcnt) for valid rows.
__global__ __launch_bounds__(256) void k_tconv(const unsigned short* y1, const unsigned short* wtp,
                                               const float* bt3, const float* bt7, const float* bt15,
                                               const float* bt31, const float* g2, const float* be2,
                                               const float* m2, const float* v2, unsigned short* xc) {
  __shared__ unsigned short Ys[158 * 68];   // up to 128+30 halo rows x 64 ic, padded (8B-aligned rows)
  const int tid = threadIdx.x, lane = tid & 31, w = tid >> 5;
  const int br = blockIdx.y;
  const int ntaps = (br == 0) ? 3 : (br == 1) ? 7 : (br == 2) ? 15 : 31;
  const int cumt  = (br == 0) ? 0 : (br == 1) ? 3 : (br == 2) ? 10 : 25;
  const float* bt = (br == 0) ? bt3 : (br == 1) ? bt7 : (br == 2) ? bt15 : bt31;
  const int pad = ntaps >> 1;
  const int b = blockIdx.x >> 5, t0 = (blockIdx.x & 31) * 128;
  const int rows = 128 + ntaps - 1;
  // stage: 8-byte chunks (4 bf16), 16 chunks per row; async DMA for in-range rows, zero halo rows
  for (int i = tid; i < rows * 16; i += 256) {
    int r = i >> 4, cg = (i & 15) * 4;
    int t = t0 - pad + r;
    unsigned short* q = Ys + r * 68 + cg;
    if (t >= 0 && t < 4096) {
      const unsigned short* g = y1 + (b * 4096 + t) * 64 + cg;
#if defined(__gfx1250__)
      unsigned loff = (unsigned)(unsigned long long)(const void*)q;  // LDS byte offset
      asm volatile("global_load_async_to_lds_b64 %0, %1, off"
                   :: "v"(loff), "v"(g)
                   : "memory");
#else
      q[0] = g[0]; q[1] = g[1]; q[2] = g[2]; q[3] = g[3];
#endif
    } else {
      q[0] = 0; q[1] = 0; q[2] = 0; q[3] = 0;
    }
  }
#if defined(__gfx1250__)
  asm volatile("s_wait_asynccnt 0x0" ::: "memory");
#endif
  __syncthreads();
  v8f acc[4];
#pragma unroll
  for (int nt = 0; nt < 4; ++nt) acc[nt] = zero8();
  const int m0 = w * 16;
  for (int tap = 0; tap < ntaps; ++tap) {
    const unsigned short* wb = wtp + ((cumt + tap) * 64) * 64;
    if (tap + 1 < ntaps) {
      // keep next tap's 8KB weight tile streaming into cache ahead of the WMMA loop
      __builtin_prefetch(wb + 4096 + lane * 128, 0, 0);
    }
#pragma unroll
    for (int ks = 0; ks < 2; ++ks) {
      Frag a;
      load_a(a, Ys + (m0 + tap) * 68 + ks * 32, 68, lane);
#pragma unroll
      for (int nt = 0; nt < 4; ++nt) {
        Frag bfr;
        load_b(bfr, wb + (nt * 16) * 64 + ks * 32, 64, lane);
        acc[nt] = wmma_bf16(a, bfr, acc[nt]);
      }
    }
  }
  const int n = lane & 15, hh = lane >> 4;
#pragma unroll
  for (int nt = 0; nt < 4; ++nt) {
    int oc = nt * 16 + n, ocg = br * 64 + oc;
    float s2 = g2[ocg] * rsqrtf(v2[ocg] + 1e-5f);
    float sh2 = be2[ocg] - m2[ocg] * s2;
    float bb = bt[oc];
#pragma unroll
    for (int r = 0; r < 8; ++r) {
      int t = t0 + m0 + r + 8 * hh;
      float v = acc[nt][r] + bb;
      v = (v > 0.f ? v : 0.f) * s2 + sh2;
      xc[(b * 4096 + t) * 256 + ocg] = f2bf(v);
    }
  }
}

// ---------------- 3) windowed DFT band power : one workgroup per (b,win) ----------------
// Y(100x256) = F(100x256) @ X(256x256); P_j = Y[2j]^2 + Y[2j+1]^2; band means -> h
__global__ __launch_bounds__(512) void k_dft(const unsigned short* xc, const unsigned short* F,
                                             unsigned short* h) {
  __shared__ __align__(16) char lds[53248];        // Xs(17408B) then reused as Ps(52000B)
  unsigned short* Xs = (unsigned short*)lds;       // [channel 256][k 32+pad]
  float* Ps = (float*)lds;                         // [jj 50][channel, stride 260]
  const int tid = threadIdx.x, lane = tid & 31, w = tid >> 5;
  const int bw = blockIdx.x, b = bw / 31, win = bw - b * 31;
  const int t0 = win * 128, n0 = w * 16;
  v8f acc[7];
#pragma unroll
  for (int mt = 0; mt < 7; ++mt) acc[mt] = zero8();
  for (int kk = 0; kk < 256; kk += 32) {
    __syncthreads();
    for (int i = tid; i < 8192; i += 512) {
      int r = i >> 8, c = i & 255;
      Xs[c * 34 + r] = xc[(b * 4096 + t0 + kk + r) * 256 + c];   // transpose to [c][k]
    }
    __syncthreads();
    Frag bf;
    load_b(bf, Xs + n0 * 34, 34, lane);
#pragma unroll
    for (int mt = 0; mt < 7; ++mt) {
      Frag a;
      load_a(a, F + (mt * 16) * 256 + kk, 256, lane);
      acc[mt] = wmma_bf16(a, bf, acc[mt]);
    }
  }
  __syncthreads();
  const int n = lane & 15, hh = lane >> 4, c = n0 + n;
#pragma unroll
  for (int mt = 0; mt < 7; ++mt) {
#pragma unroll
    for (int j = 0; j < 4; ++j) {
      int jj = mt * 8 + hh * 4 + j;                // freq index (bin k = jj+1)
      if (jj < 50) {
        float re = acc[mt][2 * j], im = acc[mt][2 * j + 1];
        Ps[jj * 260 + c] = re * re + im * im;
      }
    }
  }
  __syncthreads();
  const int bs[5] = {0, 3, 7, 12, 29};
  const int be_[5] = {4, 8, 13, 30, 50};
  for (int o = tid; o < 1280; o += 512) {
    int band = o >> 8, cc = o & 255;
    float s = 0.f;
    for (int j = bs[band]; j < be_[band]; ++j) s += Ps[j * 260 + cc];
    s *= 1.0f / (float)(be_[band] - bs[band]);
    h[bw * 1280 + band * 256 + cc] = f2bf(s);
  }
}

// ---------------- 4) spectral MLP ----------------
__global__ __launch_bounds__(256) void k_mlp1(const unsigned short* h, const unsigned short* wp1t,
                                              const float* bp1, unsigned short* hid) {
  const int tid = threadIdx.x, lane = tid & 31, w = tid >> 5;
  const int row0 = blockIdx.x * 128 + w * 16;
  v8f acc[8];
#pragma unroll
  for (int nt = 0; nt < 8; ++nt) acc[nt] = zero8();
  gemm_acc<8>(acc, h + row0 * 1280, 1280, wp1t, 1280, 40, lane);
  const int n = lane & 15, hh = lane >> 4;
#pragma unroll
  for (int nt = 0; nt < 8; ++nt) {
    int oc = nt * 16 + n;
    float bb = bp1[oc];
#pragma unroll
    for (int r = 0; r < 8; ++r) {
      float v = acc[nt][r] + bb;
      hid[(row0 + r + 8 * hh) * 128 + oc] = f2bf(v > 0.f ? v : 0.f);
    }
  }
}

__global__ __launch_bounds__(256) void k_mlp2(const unsigned short* hid, const unsigned short* wp2t,
                                              const float* bp2, float* spec) {
  const int tid = threadIdx.x, lane = tid & 31, w = tid >> 5;
  const int row0 = blockIdx.x * 128 + w * 16;
  const int n = lane & 15, hh = lane >> 4;
  for (int half = 0; half < 2; ++half) {
    v8f acc[8];
#pragma unroll
    for (int nt = 0; nt < 8; ++nt) acc[nt] = zero8();
    gemm_acc<8>(acc, hid + row0 * 128, 128, wp2t + (half * 128) * 128, 128, 4, lane);
#pragma unroll
    for (int nt = 0; nt < 8; ++nt) {
      int ng = half * 128 + nt * 16 + n;
      float bb = bp2[ng];
#pragma unroll
      for (int r = 0; r < 8; ++r) spec[(row0 + r + 8 * hh) * 256 + ng] = acc[nt][r] + bb;
    }
  }
}

// ---------------- 5) pooled xc + pooled interpolated spec -> fused-GEMM A matrix ----------------
__global__ __launch_bounds__(256) void k_pool(const unsigned short* xc, const float* spec,
                                              unsigned short* afu) {
  const int row = blockIdx.x;                    // b*128 + l
  const int b = row >> 7, l = row & 127, c = threadIdx.x;
  float sx = 0.f, ss = 0.f;
  for (int j = 0; j < 32; ++j) {
    int t = l * 32 + j;
    sx += bf2f(xc[(b * 4096 + t) * 256 + c]);
    float src = ((float)t + 0.5f) * (31.0f / 4096.0f) - 0.5f;
    src = fminf(fmaxf(src, 0.f), 30.f);
    int i0 = (int)floorf(src);
    int i1 = (i0 + 1 < 30) ? i0 + 1 : 30;
    float wg = src - (float)i0;
    ss += spec[(b * 31 + i0) * 256 + c] * (1.f - wg) + spec[(b * 31 + i1) * 256 + c] * wg;
  }
  afu[row * 512 + c]       = f2bf(sx * (1.f / 32.f));
  afu[row * 512 + 256 + c] = f2bf(ss * (1.f / 32.f));
}

// ---------------- 6) fused matmul (pooled) + b_fu + pos -> pp bf16 ----------------
__global__ __launch_bounds__(256) void k_fuse(const unsigned short* afu, const unsigned short* wfut,
                                              const float* bfu, const float* pos, unsigned short* pp) {
  const int tid = threadIdx.x, lane = tid & 31, w = tid >> 5;
  const int row0 = blockIdx.x * 128 + w * 16;
  const int n = lane & 15, hh = lane >> 4;
  for (int half = 0; half < 2; ++half) {
    v8f acc[8];
#pragma unroll
    for (int nt = 0; nt < 8; ++nt) acc[nt] = zero8();
    gemm_acc<8>(acc, afu + row0 * 512, 512, wfut + (half * 128) * 512, 512, 16, lane);
#pragma unroll
    for (int nt = 0; nt < 8; ++nt) {
      int ng = half * 128 + nt * 16 + n;
      float bb = bfu[ng];
#pragma unroll
      for (int r = 0; r < 8; ++r) {
        int rw = row0 + r + 8 * hh;
        int l = rw & 127;
        pp[rw * 256 + ng] = f2bf(acc[nt][r] + bb + pos[l * 256 + ng]);
      }
    }
  }
}

// ---------------- 7) output projection ----------------
__global__ __launch_bounds__(256) void k_out(const unsigned short* pp, const unsigned short* woutt,
                                             const float* bout, float* out) {
  const int tid = threadIdx.x, lane = tid & 31, w = tid >> 5;
  const int row0 = blockIdx.x * 128 + w * 16;
  const int n = lane & 15, hh = lane >> 4;
  for (int half = 0; half < 2; ++half) {
    v8f acc[8];
#pragma unroll
    for (int nt = 0; nt < 8; ++nt) acc[nt] = zero8();
    gemm_acc<8>(acc, pp + row0 * 256, 256, woutt + (half * 128) * 256, 256, 8, lane);
#pragma unroll
    for (int nt = 0; nt < 8; ++nt) {
      int ng = half * 128 + nt * 16 + n;
      float bb = bout[ng];
#pragma unroll
      for (int r = 0; r < 8; ++r) out[(row0 + r + 8 * hh) * 256 + ng] = acc[nt][r] + bb;
    }
  }
}

// ---------------- host launch ----------------
static inline size_t align256(size_t x) { return (x + 255) & ~(size_t)255; }

extern "C" void kernel_launch(void* const* d_in, const int* in_sizes, int n_in,
                              void* d_out, int out_size, void* d_ws, size_t ws_size,
                              hipStream_t stream) {
  const float* x     = (const float*)d_in[0];
  const float* w_sp  = (const float*)d_in[1];
  const float* b_sp  = (const float*)d_in[2];
  const float* g1    = (const float*)d_in[3];
  const float* be1   = (const float*)d_in[4];
  const float* m1    = (const float*)d_in[5];
  const float* v1    = (const float*)d_in[6];
  const float* wt3   = (const float*)d_in[7];
  const float* bt3   = (const float*)d_in[8];
  const float* wt7   = (const float*)d_in[9];
  const float* bt7   = (const float*)d_in[10];
  const float* wt15  = (const float*)d_in[11];
  const float* bt15  = (const float*)d_in[12];
  const float* wt31  = (const float*)d_in[13];
  const float* bt31  = (const float*)d_in[14];
  const float* g2    = (const float*)d_in[15];
  const float* be2   = (const float*)d_in[16];
  const float* m2    = (const float*)d_in[17];
  const float* v2    = (const float*)d_in[18];
  const float* wp1   = (const float*)d_in[19];
  const float* bp1   = (const float*)d_in[20];
  const float* wp2   = (const float*)d_in[21];
  const float* bp2   = (const float*)d_in[22];
  const float* w_fu  = (const float*)d_in[23];
  const float* b_fu  = (const float*)d_in[24];
  const float* pos   = (const float*)d_in[25];
  const float* w_out = (const float*)d_in[26];
  const float* b_out = (const float*)d_in[27];
  float* out = (float*)d_out;

  char* ws = (char*)d_ws;
  size_t off = 0;
  unsigned short* Y1   = (unsigned short*)(ws + off); off = align256(off + (size_t)131072 * 64 * 2);
  unsigned short* XC   = (unsigned short*)(ws + off); off = align256(off + (size_t)131072 * 256 * 2);
  unsigned short* H    = (unsigned short*)(ws + off); off = align256(off + (size_t)1024 * 1280 * 2);
  unsigned short* HID  = (unsigned short*)(ws + off); off = align256(off + (size_t)1024 * 128 * 2);
  float*          SPEC = (float*)(ws + off);          off = align256(off + (size_t)1024 * 256 * 4);
  unsigned short* AFU  = (unsigned short*)(ws + off); off = align256(off + (size_t)4096 * 512 * 2);
  unsigned short* PP   = (unsigned short*)(ws + off); off = align256(off + (size_t)4096 * 256 * 2);
  unsigned short* FM   = (unsigned short*)(ws + off); off = align256(off + (size_t)112 * 256 * 2);
  unsigned short* W1P  = (unsigned short*)(ws + off); off = align256(off + (size_t)64 * 256 * 2);
  float*          B1F  = (float*)(ws + off);          off = align256(off + (size_t)64 * 4);
  unsigned short* WTP  = (unsigned short*)(ws + off); off = align256(off + (size_t)56 * 64 * 64 * 2);
  unsigned short* WP1T = (unsigned short*)(ws + off); off = align256(off + (size_t)128 * 1280 * 2);
  unsigned short* WP2T = (unsigned short*)(ws + off); off = align256(off + (size_t)256 * 128 * 2);
  unsigned short* WFUT = (unsigned short*)(ws + off); off = align256(off + (size_t)256 * 512 * 2);
  unsigned short* WOUT = (unsigned short*)(ws + off); off = align256(off + (size_t)256 * 256 * 2);
  (void)ws_size; (void)n_in; (void)in_sizes; (void)out_size;

  // --- prep (cheap, every call; deterministic) ---
  k_prep_pw<<<64, 256, 0, stream>>>(w_sp, b_sp, g1, be1, m1, v1, W1P, B1F);
  k_prep_wt<<<896, 256, 0, stream>>>(wt3, wt7, wt15, wt31, WTP);
  k_prep_F<<<112, 256, 0, stream>>>(FM);
  k_tpack<<<640, 256, 0, stream>>>(wp1, WP1T, 1280, 128);
  k_tpack<<<128, 256, 0, stream>>>(wp2, WP2T, 128, 256);
  k_tpack<<<512, 256, 0, stream>>>(w_fu, WFUT, 512, 256);
  k_tpack<<<256, 256, 0, stream>>>(w_out, WOUT, 256, 256);
  k_zero_hpad<<<160, 256, 0, stream>>>(H);

  // --- pipeline ---
  k_pointwise<<<1024, 256, 0, stream>>>(x, W1P, B1F, Y1);
  k_tconv<<<dim3(1024, 4), 256, 0, stream>>>(Y1, WTP, bt3, bt7, bt15, bt31, g2, be2, m2, v2, XC);
  k_dft<<<992, 512, 0, stream>>>(XC, FM, H);
  k_mlp1<<<8, 256, 0, stream>>>(H, WP1T, bp1, HID);
  k_mlp2<<<8, 256, 0, stream>>>(HID, WP2T, bp2, SPEC);
  k_pool<<<4096, 256, 0, stream>>>(XC, SPEC, AFU);
  k_fuse<<<32, 256, 0, stream>>>(AFU, WFUT, b_fu, pos, PP);
  k_out<<<32, 256, 0, stream>>>(PP, WOUT, b_out, out);
}
